// Attention_84361747628350
// MI455X (gfx1250) — matrix operand order repaired and checked by reading, not compiled, not run on hardware
//
#include <hip/hip_runtime.h>
#include <math.h>

// ---------------- CDNA5 types ----------------
typedef __attribute__((ext_vector_type(16))) __bf16   v16bf;
typedef __attribute__((ext_vector_type(8)))  float    v8f;
typedef __attribute__((ext_vector_type(4)))  unsigned u32x4;

union FragAB { u32x4 u[2]; v16bf v; };   // 32 bytes = 8 VGPRs: one WMMA A/B operand

#define D_MODEL 2048
#define SEQ     2048
#define NHEADS  16
#define HD      128
#define BATCH   2
#define MTOT    (BATCH * SEQ)   // 4096 rows in the projection GEMMs

// f32 -> bf16 with round-to-nearest-even
__device__ __forceinline__ unsigned short f2bf(float f) {
  unsigned u = __builtin_bit_cast(unsigned, f);
  u += 0x7FFFu + ((u >> 16) & 1u);
  return (unsigned short)(u >> 16);
}

// Row reductions across the 16-lane half of a wave32 (C-matrix rows live in 16 lanes)
__device__ __forceinline__ float hmax16(float v) {
#pragma unroll
  for (int m = 1; m < 16; m <<= 1) v = fmaxf(v, __shfl_xor(v, m, 32));
  return v;
}
__device__ __forceinline__ float hsum16(float v) {
#pragma unroll
  for (int m = 1; m < 16; m <<= 1) v += __shfl_xor(v, m, 32);
  return v;
}

// CDNA5 async global->LDS copy (GV mode): writes LDS directly, tracked by ASYNCcnt.
// LDS offset = low 32 bits of the generic shared-space address (ISA 10.2 aperture rule).
__device__ __forceinline__ void async_copy_b128(void* lds_ptr, const void* gptr) {
  unsigned lds_off = (unsigned)(size_t)lds_ptr;
  unsigned long long ga = (unsigned long long)gptr;
  asm volatile("global_load_async_to_lds_b128 %0, %1, off"
               :: "v"(lds_off), "v"(ga) : "memory");
}
__device__ __forceinline__ void wait_asynccnt0() {
  asm volatile("s_wait_asynccnt 0x0" ::: "memory");
}

// ---------------- Kernel 0: f32 -> bf16 conversion pass ----------------
__global__ __launch_bounds__(256) void cvt_bf16(const float* __restrict__ in,
                                                unsigned short* __restrict__ out, int n) {
  for (int i = blockIdx.x * blockDim.x + threadIdx.x; i < n; i += gridDim.x * blockDim.x)
    out[i] = f2bf(in[i]);
}

// ---------------- Kernel 1: QKV projection GEMM + RoPE epilogue ----------------
// C[M=4096, N=2048] = X_bf16 @ W^T, W stored [N, K] row-major (so both operands are
// contiguous along K). blockIdx.z selects wq/wk/wv. Q,K get RoPE and land in
// [B,H,S,hd]; V lands transposed in [B,H,hd,S] for the flash kernel's B-operand.
__global__ __launch_bounds__(256) void qkv_gemm_rope(
    const unsigned short* __restrict__ xb,
    const unsigned short* __restrict__ wqb,
    const unsigned short* __restrict__ wkb,
    const unsigned short* __restrict__ wvb,
    const float* __restrict__ fcos,
    const float* __restrict__ fsin,
    unsigned short* __restrict__ Qb,
    unsigned short* __restrict__ Kb,
    unsigned short* __restrict__ Vt) {
  __shared__ __align__(16) unsigned short As[128][32];
  __shared__ __align__(16) unsigned short Bs[128][32];

  const int tid  = threadIdx.x;
  const int lane = tid & 31, wave = tid >> 5;
  const int m16  = lane & 15, hl = lane >> 4;
  const int wm   = wave & 1,  wn = wave >> 1;          // 2x4 wave grid -> 128x128 block
  const int blockN = blockIdx.x * 128, blockM = blockIdx.y * 128;
  const int z = blockIdx.z;
  const unsigned short* Wg = (z == 0) ? wqb : (z == 1) ? wkb : wvb;

  v8f acc[4][2] = {};
  const int k0a = hl * 8;    // A-operand K split {0-7,16-23} / {8-15,24-31}
  const int kb0 = hl * 8;    // B-operand K split {0-7,16-23} / {8-15,24-31}, same as A

  for (int kt = 0; kt < D_MODEL; kt += 32) {
    // Async global->LDS tile fill: 256 threads x 2 x 16B per array (8KB each)
#pragma unroll
    for (int i = 0; i < 2; i++) {
      int c = tid + i * 256;
      int r = c >> 2, col = (c & 3) * 8;
      async_copy_b128(&As[r][col], &xb[(size_t)(blockM + r) * D_MODEL + kt + col]);
      async_copy_b128(&Bs[r][col], &Wg[(size_t)(blockN + r) * D_MODEL + kt + col]);
    }
    wait_asynccnt0();
    __syncthreads();
    if (kt + 32 < D_MODEL) {   // global_prefetch_b8 for the next K tile
      int r0 = tid >> 1;
      __builtin_prefetch(&xb[(size_t)(blockM + r0) * D_MODEL + kt + 32], 0, 1);
      __builtin_prefetch(&Wg[(size_t)(blockN + r0) * D_MODEL + kt + 32], 0, 1);
    }
    FragAB fa[4], fb[2];
#pragma unroll
    for (int msb = 0; msb < 4; msb++) {
      const unsigned short* p = &As[wm * 64 + msb * 16 + m16][k0a];
      fa[msb].u[0] = *(const u32x4*)p;
      fa[msb].u[1] = *(const u32x4*)(p + 16);
    }
#pragma unroll
    for (int nsb = 0; nsb < 2; nsb++) {
      const unsigned short* p = &Bs[wn * 32 + nsb * 16 + m16][kb0];
      fb[nsb].u[0] = *(const u32x4*)p;
      fb[nsb].u[1] = *(const u32x4*)(p + 16);
    }
#pragma unroll
    for (int msb = 0; msb < 4; msb++)
#pragma unroll
      for (int nsb = 0; nsb < 2; nsb++)
        acc[msb][nsb] = __builtin_amdgcn_wmma_f32_16x16x32_bf16(
            false, fa[msb].v, false, fb[nsb].v, (short)0, acc[msb][nsb], false, false);
    __syncthreads();
  }

  // Epilogue: C layout => row = r + 8*hl (per 16x16 tile), col = lane&15
#pragma unroll
  for (int msb = 0; msb < 4; msb++) {
#pragma unroll
    for (int nsb = 0; nsb < 2; nsb++) {
#pragma unroll
      for (int r = 0; r < 8; r++) {
        int m = blockM + wm * 64 + msb * 16 + r + hl * 8;
        int n = blockN + wn * 32 + nsb * 16 + m16;
        float v = acc[msb][nsb][r];
        int b = m >> 11, s = m & (SEQ - 1);
        int h = n >> 7,  d = n & (HD - 1);
        if (z == 2) {
          // V stored transposed: [B,H,hd,S]
          Vt[((size_t)((b * NHEADS + h) * HD + d)) * SEQ + s] = f2bf(v);
        } else {
          // RoPE: pair (2i,2i+1) sits in adjacent lanes of this half-wave
          float p = __shfl_xor(v, 1, 32);
          int fi = s * (HD / 2) + (d >> 1);
          float fc = fcos[fi], fs = fsin[fi];
          float res = (d & 1) ? (p * fs + v * fc) : (v * fc - p * fs);
          size_t idx = ((size_t)((b * NHEADS + h) * SEQ + s)) * HD + d;
          if (z == 0) Qb[idx] = f2bf(res); else Kb[idx] = f2bf(res);
        }
      }
    }
  }
}

// ---------------- Kernel 2: causal flash attention (WMMA QK^T and PV) ----------------
// One 16-row q-tile per wave (8 independent waves per block), kv streamed in tiles
// of 32 with online softmax; P goes C-layout -> A-layout through per-wave LDS.
__global__ __launch_bounds__(256) void flash_attn(
    const unsigned short* __restrict__ Qb,
    const unsigned short* __restrict__ Kb,
    const unsigned short* __restrict__ Vt,
    unsigned short* __restrict__ Ob) {
  __shared__ __align__(16) unsigned short Psh[8][16 * 32];

  const int lane = threadIdx.x & 31, wave = threadIdx.x >> 5;
  const int m16 = lane & 15, hl = lane >> 4;
  const int q0 = blockIdx.x * 128 + wave * 16;
  const int bh = blockIdx.y;
  const int b = bh >> 4, h = bh & 15;

  const unsigned short* Qp = Qb + (size_t)bh * SEQ * HD;
  const unsigned short* Kp = Kb + (size_t)bh * SEQ * HD;
  const unsigned short* Vp = Vt + (size_t)bh * HD * SEQ;

  // Q tile 16x128 as four A-operands (resident for the whole kv loop)
  FragAB qf[4];
  {
    const int qrow = q0 + m16, k0 = hl * 8;
#pragma unroll
    for (int c = 0; c < 4; c++) {
      const unsigned short* p = &Qp[(size_t)qrow * HD + c * 32 + k0];
      qf[c].u[0] = *(const u32x4*)p;
      qf[c].u[1] = *(const u32x4*)(p + 16);
    }
  }

  float mrow[8], lrow[8];
  v8f O[8] = {};
#pragma unroll
  for (int r = 0; r < 8; r++) { mrow[r] = -1e30f; lrow[r] = 0.f; }

  const float scale = 0.088388347648318447f;  // 1/sqrt(hd)
  unsigned short* Pw = &Psh[wave][0];

  for (int kt = 0; kt <= q0 + 15; kt += 32) {   // causal early termination
    // scores: S(16x32) = Q(16x128) x K^T(128x32), two 16x16 accumulators
    v8f sacc[2] = {};
#pragma unroll
    for (int ns = 0; ns < 2; ns++) {
      const int kvrow = kt + ns * 16 + m16;     // B-operand: column = lane&15
#pragma unroll
      for (int c = 0; c < 4; c++) {
        FragAB kf;
        const unsigned short* p = &Kp[(size_t)kvrow * HD + c * 32 + hl * 8];
        kf.u[0] = *(const u32x4*)p;
        kf.u[1] = *(const u32x4*)(p + 16);
        sacc[ns] = __builtin_amdgcn_wmma_f32_16x16x32_bf16(
            false, qf[c].v, false, kf.v, (short)0, sacc[ns], false, false);
      }
    }
    // online softmax (row stats replicated across the 16-lane half)
    float alpha[8];
#pragma unroll
    for (int r = 0; r < 8; r++) {
      const int qr = q0 + r + hl * 8;
      float s0 = (kt + m16      <= qr) ? sacc[0][r] * scale : -1e30f;
      float s1 = (kt + 16 + m16 <= qr) ? sacc[1][r] * scale : -1e30f;
      float mnew = fmaxf(mrow[r], hmax16(fmaxf(s0, s1)));
      float a  = __expf(mrow[r] - mnew);
      float p0 = __expf(s0 - mnew);
      float p1 = __expf(s1 - mnew);
      lrow[r] = lrow[r] * a + hsum16(p0 + p1);
      mrow[r] = mnew;
      alpha[r] = a;
      // P in C layout -> per-wave LDS scratch (same-wave DS ops stay in order)
      Pw[(r + hl * 8) * 32 + m16]      = f2bf(p0);
      Pw[(r + hl * 8) * 32 + 16 + m16] = f2bf(p1);
    }
#pragma unroll
    for (int t = 0; t < 8; t++)
#pragma unroll
      for (int r = 0; r < 8; r++) O[t][r] *= alpha[r];

    // reload P in A-operand layout
    FragAB pf;
    pf.u[0] = *(const u32x4*)&Pw[m16 * 32 + hl * 8];
    pf.u[1] = *(const u32x4*)&Pw[m16 * 32 + hl * 8 + 16];

    // O(16x128) += P(16x32) x V(32x128); V^T layout makes B-operand contiguous
#pragma unroll
    for (int t = 0; t < 8; t++) {
      FragAB vf;
      const unsigned short* p = &Vp[(size_t)(t * 16 + m16) * SEQ + kt + hl * 8];
      vf.u[0] = *(const u32x4*)p;
      vf.u[1] = *(const u32x4*)(p + 16);
      O[t] = __builtin_amdgcn_wmma_f32_16x16x32_bf16(
          false, pf.v, false, vf.v, (short)0, O[t], false, false);
    }
  }

  // normalize and store attn output as [B,S,H,hd] == [B,S,D] bf16
#pragma unroll
  for (int t = 0; t < 8; t++) {
#pragma unroll
    for (int r = 0; r < 8; r++) {
      const int srow = q0 + r + hl * 8;
      float o = O[t][r] / lrow[r];
      Ob[((size_t)((b * SEQ + srow) * NHEADS + h)) * HD + t * 16 + m16] = f2bf(o);
    }
  }
}

// ---------------- Kernel 3: output projection GEMM (f32 result) ----------------
__global__ __launch_bounds__(256) void out_gemm(
    const unsigned short* __restrict__ Ab,
    const unsigned short* __restrict__ Wb,
    float* __restrict__ out) {
  __shared__ __align__(16) unsigned short As[128][32];
  __shared__ __align__(16) unsigned short Bs[128][32];

  const int tid  = threadIdx.x;
  const int lane = tid & 31, wave = tid >> 5;
  const int m16  = lane & 15, hl = lane >> 4;
  const int wm   = wave & 1,  wn = wave >> 1;
  const int blockN = blockIdx.x * 128, blockM = blockIdx.y * 128;

  v8f acc[4][2] = {};
  const int k0a = hl * 8;
  const int kb0 = hl * 8;

  for (int kt = 0; kt < D_MODEL; kt += 32) {
#pragma unroll
    for (int i = 0; i < 2; i++) {
      int c = tid + i * 256;
      int r = c >> 2, col = (c & 3) * 8;
      async_copy_b128(&As[r][col], &Ab[(size_t)(blockM + r) * D_MODEL + kt + col]);
      async_copy_b128(&Bs[r][col], &Wb[(size_t)(blockN + r) * D_MODEL + kt + col]);
    }
    wait_asynccnt0();
    __syncthreads();
    if (kt + 32 < D_MODEL) {
      int r0 = tid >> 1;
      __builtin_prefetch(&Ab[(size_t)(blockM + r0) * D_MODEL + kt + 32], 0, 1);
      __builtin_prefetch(&Wb[(size_t)(blockN + r0) * D_MODEL + kt + 32], 0, 1);
    }
    FragAB fa[4], fb[2];
#pragma unroll
    for (int msb = 0; msb < 4; msb++) {
      const unsigned short* p = &As[wm * 64 + msb * 16 + m16][k0a];
      fa[msb].u[0] = *(const u32x4*)p;
      fa[msb].u[1] = *(const u32x4*)(p + 16);
    }
#pragma unroll
    for (int nsb = 0; nsb < 2; nsb++) {
      const unsigned short* p = &Bs[wn * 32 + nsb * 16 + m16][kb0];
      fb[nsb].u[0] = *(const u32x4*)p;
      fb[nsb].u[1] = *(const u32x4*)(p + 16);
    }
#pragma unroll
    for (int msb = 0; msb < 4; msb++)
#pragma unroll
      for (int nsb = 0; nsb < 2; nsb++)
        acc[msb][nsb] = __builtin_amdgcn_wmma_f32_16x16x32_bf16(
            false, fa[msb].v, false, fb[nsb].v, (short)0, acc[msb][nsb], false, false);
    __syncthreads();
  }

#pragma unroll
  for (int msb = 0; msb < 4; msb++)
#pragma unroll
    for (int nsb = 0; nsb < 2; nsb++)
#pragma unroll
      for (int r = 0; r < 8; r++) {
        int m = blockM + wm * 64 + msb * 16 + r + hl * 8;
        int n = blockN + wn * 32 + nsb * 16 + m16;
        out[(size_t)m * D_MODEL + n] = acc[msb][nsb][r];
      }
}

// ---------------- Host side ----------------
extern "C" void kernel_launch(void* const* d_in, const int* in_sizes, int n_in,
                              void* d_out, int out_size, void* d_ws, size_t ws_size,
                              hipStream_t stream) {
  (void)in_sizes; (void)n_in; (void)out_size; (void)ws_size;

  const float* x    = (const float*)d_in[0];
  const float* fcos = (const float*)d_in[1];
  const float* fsin = (const float*)d_in[2];
  const float* wq   = (const float*)d_in[3];
  const float* wk   = (const float*)d_in[4];
  const float* wv   = (const float*)d_in[5];
  const float* wo   = (const float*)d_in[6];

  const size_t NX = (size_t)BATCH * SEQ * D_MODEL;  // 8.39M elems
  const size_t NW = (size_t)D_MODEL * D_MODEL;      // 4.19M elems

  // Workspace layout (bf16 ushorts): ~117 MB total
  unsigned short* xb  = (unsigned short*)d_ws;
  unsigned short* wqb = xb  + NX;
  unsigned short* wkb = wqb + NW;
  unsigned short* wvb = wkb + NW;
  unsigned short* wob = wvb + NW;
  unsigned short* Qb  = wob + NW;   // [B,H,S,hd]
  unsigned short* Kb  = Qb  + NX;   // [B,H,S,hd]
  unsigned short* Vt  = Kb  + NX;   // [B,H,hd,S] (transposed)
  unsigned short* Ob  = Vt  + NX;   // [B,S,D]

  cvt_bf16<<<2048, 256, 0, stream>>>(x,  xb,  (int)NX);
  cvt_bf16<<<1024, 256, 0, stream>>>(wq, wqb, (int)NW);
  cvt_bf16<<<1024, 256, 0, stream>>>(wk, wkb, (int)NW);
  cvt_bf16<<<1024, 256, 0, stream>>>(wv, wvb, (int)NW);
  cvt_bf16<<<1024, 256, 0, stream>>>(wo, wob, (int)NW);

  dim3 g1(D_MODEL / 128, MTOT / 128, 3);       // 16 x 32 x 3
  qkv_gemm_rope<<<g1, 256, 0, stream>>>(xb, wqb, wkb, wvb, fcos, fsin, Qb, Kb, Vt);

  dim3 g2(SEQ / 128, BATCH * NHEADS);          // 16 x 32
  flash_attn<<<g2, 256, 0, stream>>>(Qb, Kb, Vt, Ob);

  dim3 g3(D_MODEL / 128, MTOT / 128);          // 16 x 32
  out_gemm<<<g3, 256, 0, stream>>>(Ob, wob, (float*)d_out);
}
